// GnosisAttention_1735166787889
// MI455X (gfx1250) — matrix operand order repaired and checked by reading, not compiled, not run on hardware
//
#include <hip/hip_runtime.h>
#include <hip/hip_bf16.h>

typedef __attribute__((ext_vector_type(16))) __bf16 v16bf;
typedef __attribute__((ext_vector_type(8)))  __bf16 v8bf;
typedef __attribute__((ext_vector_type(8)))  float  v8f;

#define D_MODEL 1024
#define G_DIM   64
#define BATCH   4
#define SEQ     2048
#define QROWS   32     // query rows per workgroup (2 WMMA M-tiles)
#define KT      128    // keys per tile iteration
#define NWAVES  8      // waves per block (256 threads, wave32)

// ---------------------------------------------------------------------------
// WMMA fragment loaders (layouts per CDNA5 ISA 7.12.2, wave32)
// A-matrix 16x32 bf16, row-major source (M x K):
//   lanes 0-15: row M=lane, K = 0..7 (vgpr0-3) and 16..23 (vgpr4-7)
//   lanes16-31: row M=lane-16, K = 8..15 and 24..31
__device__ __forceinline__ v16bf load_afrag(const __bf16* p, int ld, int lane) {
    const int m  = lane & 15;
    const int ko = (lane >> 4) << 3;         // 0 or 8
    const __bf16* q = p + (long long)m * ld + ko;
    v8bf lo = *(const v8bf*)(q);             // K = ko .. ko+7
    v8bf hi = *(const v8bf*)(q + 16);        // K = ko+16 .. ko+23
    v16bf r;
#pragma unroll
    for (int i = 0; i < 8; ++i) { r[i] = lo[i]; r[i + 8] = hi[i]; }
    return r;
}

// B-matrix 32x16 bf16 from a row-major (N x K) buffer (row n == column n of B):
//   lanes 0-15: col N=lane,    K = 0..7 and 16..23
//   lanes16-31: col N=lane-16, K = 8..15 and 24..31
__device__ __forceinline__ v16bf load_bfrag(const __bf16* p, int ld, int lane) {
    const int n  = lane & 15;
    const int ko = (lane >> 4) << 3;         // 0 or 8
    const __bf16* q = p + (long long)n * ld + ko;
    v8bf lo = *(const v8bf*)(q);
    v8bf hi = *(const v8bf*)(q + 16);
    v16bf r;
#pragma unroll
    for (int i = 0; i < 8; ++i) { r[i] = lo[i]; r[i + 8] = hi[i]; }
    return r;
}

__device__ __forceinline__ v8f wmma_bf16(v16bf a, v16bf b, v8f c) {
    return __builtin_amdgcn_wmma_f32_16x16x32_bf16(false, a, false, b,
                                                   (short)0, c, false, false);
}

__device__ __forceinline__ v8f v8f_zero() {
    v8f z;
#pragma unroll
    for (int i = 0; i < 8; ++i) z[i] = 0.0f;
    return z;
}

// ---------------------------------------------------------------------------
// fp32 -> bf16 convert (grid-stride)
__global__ void k_cvt_bf16(const float* __restrict__ s, __bf16* __restrict__ d,
                           long long n) {
    long long i = (long long)blockIdx.x * blockDim.x + threadIdx.x;
    long long stride = (long long)gridDim.x * blockDim.x;
    for (; i < n; i += stride) d[i] = (__bf16)s[i];
}

// ---------------------------------------------------------------------------
// V[b][s][d] (f32) -> Vt[b][d][s] (bf16), 32x32 LDS-tiled transpose
__global__ __launch_bounds__(256) void k_transpose_v(const float* __restrict__ V,
                                                     __bf16* __restrict__ Vt) {
    const int SD_TILES = (SEQ / 32) * (D_MODEL / 32);
    int bid = blockIdx.x;
    int b  = bid / SD_TILES;
    int t  = bid % SD_TILES;
    int st = t / (D_MODEL / 32);
    int dt = t % (D_MODEL / 32);
    __shared__ float tile[32][33];
    int tx = threadIdx.x & 31;
    int ty = threadIdx.x >> 5;
    const float* src = V + ((long long)b * SEQ + st * 32) * D_MODEL + dt * 32;
#pragma unroll
    for (int i = 0; i < 4; ++i)
        tile[ty + 8 * i][tx] = src[(long long)(ty + 8 * i) * D_MODEL + tx];
    __syncthreads();
    __bf16* dst = Vt + ((long long)b * D_MODEL + dt * 32) * SEQ + st * 32;
#pragma unroll
    for (int i = 0; i < 4; ++i)
        dst[(long long)(ty + 8 * i) * SEQ + tx] = (__bf16)tile[tx][ty + 8 * i];
}

// ---------------------------------------------------------------------------
// C[m][n] = act( sum_k A[m][k]*Bm[n][k] + bias[n] )
// A: (M x K) bf16 row-major.  Bm: (N x K) bf16 row-major (i.e. B^T).
// One wave computes a 32x64 output tile (two M-tiles share every B fragment).
// ACT: 0=identity, 1=tanh.
template <int ACT, typename OutT>
__global__ __launch_bounds__(256) void k_gemm_bf16(
    const __bf16* __restrict__ A, const __bf16* __restrict__ Bm,
    const float* __restrict__ bias, OutT* __restrict__ C,
    int M, int N, int K, int ldc) {
    const int lane = threadIdx.x & 31;
    const int gw = blockIdx.x * NWAVES + (threadIdx.x >> 5);
    const int ngroups = N >> 6;
    const int mt = gw / ngroups;
    const int ng = gw - mt * ngroups;
    if (mt * 32 >= M) return;

    v8f acc[4][2];
#pragma unroll
    for (int t = 0; t < 4; ++t) { acc[t][0] = v8f_zero(); acc[t][1] = v8f_zero(); }

    const __bf16* Ap0 = A + (long long)mt * 32 * K;
    const __bf16* Ap1 = Ap0 + (long long)16 * K;
    for (int kk = 0; kk < K; kk += 32) {
        v16bf a0 = load_afrag(Ap0 + kk, K, lane);
        v16bf a1 = load_afrag(Ap1 + kk, K, lane);
#pragma unroll
        for (int t = 0; t < 4; ++t) {
            v16bf bf_ = load_bfrag(Bm + (long long)(ng * 64 + t * 16) * K + kk, K, lane);
            acc[t][0] = wmma_bf16(a0, bf_, acc[t][0]);
            acc[t][1] = wmma_bf16(a1, bf_, acc[t][1]);
        }
    }

    const int half = lane >> 4;
    const int lm   = lane & 15;
#pragma unroll
    for (int t = 0; t < 4; ++t) {
        int n = ng * 64 + t * 16 + lm;
        float bv = bias[n];
#pragma unroll
        for (int m01 = 0; m01 < 2; ++m01) {
#pragma unroll
            for (int r = 0; r < 8; ++r) {
                int m = mt * 32 + m01 * 16 + r + 8 * half;
                float v = acc[t][m01][r] + bv;
                if (ACT == 1) v = tanhf(v);
                C[(long long)m * ldc + n] = (OutT)v;
            }
        }
    }
}

// ---------------------------------------------------------------------------
// Fused gated flash attention.
// Block = 256 threads (8 waves), one block per (batch, 32-query-row tile).
// Per key tile of 128:
//   wave w: two 16x16 score tiles (keys 16w..16w+15) via WMMA over K=1024
//           sharing each K-row B fragment, gnosis via WMMA over K=64,
//           sigmoid gate, online softmax (cross-wave stats in LDS),
//           P -> LDS (bf16), then O-slice [128w,128w+128) += P @ V where each
//           V fragment feeds both M-tiles.
__global__ __launch_bounds__(256) void k_attention(
    const __bf16* __restrict__ Qb, const __bf16* __restrict__ Kb,
    const __bf16* __restrict__ Vt,
    const __bf16* __restrict__ Gq, const __bf16* __restrict__ Gk,
    const float* __restrict__ wgate_p, const float* __restrict__ bgate_p,
    __bf16* __restrict__ Ob) {
    const int b   = blockIdx.x / (SEQ / QROWS);
    const int qt  = blockIdx.x % (SEQ / QROWS);
    const int tid  = threadIdx.x;
    const int wave = tid >> 5;
    const int lane = tid & 31;
    const int half = lane >> 4;
    const int lm   = lane & 15;

    __shared__ __align__(16) __bf16 Qs[QROWS][D_MODEL + 8];  // padded rows
    __shared__ __align__(16) __bf16 Gqs[QROWS][G_DIM + 8];
    __shared__ __align__(16) __bf16 Ps[QROWS][KT + 8];
    __shared__ float wmaxs[NWAVES][QROWS];
    __shared__ float wsums[NWAVES][QROWS];
    __shared__ float run_m[QROWS];
    __shared__ float run_s[QROWS];

    const float wgate = *wgate_p;
    const float bgate = *bgate_p;

    const long long qrow0 = (long long)(b * SEQ + qt * QROWS);

    for (int i = tid; i < QROWS * D_MODEL; i += 256) {
        int r = i >> 10, c = i & (D_MODEL - 1);
        Qs[r][c] = Qb[(qrow0 + r) * D_MODEL + c];
    }
    for (int i = tid; i < QROWS * G_DIM; i += 256) {
        int r = i >> 6, c = i & (G_DIM - 1);
        Gqs[r][c] = Gq[(qrow0 + r) * G_DIM + c];
    }
    if (tid < QROWS) { run_m[tid] = -1e30f; run_s[tid] = 0.0f; }
    __syncthreads();

    v8f oacc[8][2];
#pragma unroll
    for (int t = 0; t < 8; ++t) { oacc[t][0] = v8f_zero(); oacc[t][1] = v8f_zero(); }

    const float inv_scale = 0.03125f;  // 1/sqrt(1024)

    for (int ktile = 0; ktile < SEQ / KT; ++ktile) {
        const int kbase = ktile * KT + wave * 16;

        // ---- raw score tiles (2 x 16 q-rows x 16 keys), K = D_MODEL
        v8f sacc[2] = { v8f_zero(), v8f_zero() };
        const __bf16* Krows = Kb + ((long long)(b * SEQ) + kbase) * D_MODEL;
        for (int kk = 0; kk < D_MODEL; kk += 32) {
            v16bf bf_ = load_bfrag(Krows + kk, D_MODEL, lane);
            v16bf a0  = load_afrag(&Qs[0][0] + kk, D_MODEL + 8, lane);
            v16bf a1  = load_afrag(&Qs[16][0] + kk, D_MODEL + 8, lane);
            sacc[0] = wmma_bf16(a0, bf_, sacc[0]);
            sacc[1] = wmma_bf16(a1, bf_, sacc[1]);
        }

        // ---- gnosis tiles, K = G_DIM
        v8f gacc[2] = { v8f_zero(), v8f_zero() };
        const __bf16* Gkrows = Gk + ((long long)(b * SEQ) + kbase) * G_DIM;
#pragma unroll
        for (int kk = 0; kk < G_DIM; kk += 32) {
            v16bf bf_ = load_bfrag(Gkrows + kk, G_DIM, lane);
            v16bf a0  = load_afrag(&Gqs[0][0] + kk, G_DIM + 8, lane);
            v16bf a1  = load_afrag(&Gqs[16][0] + kk, G_DIM + 8, lane);
            gacc[0] = wmma_bf16(a0, bf_, gacc[0]);
            gacc[1] = wmma_bf16(a1, bf_, gacc[1]);
        }

        // ---- gate: s = s/sqrt(d) * (1 + sigmoid(g*w + b))
        float sv[16];
#pragma unroll
        for (int m01 = 0; m01 < 2; ++m01) {
#pragma unroll
            for (int r = 0; r < 8; ++r) {
                float sig = 1.0f / (1.0f + __expf(-(gacc[m01][r] * wgate + bgate)));
                sv[m01 * 8 + r] = sacc[m01][r] * inv_scale * (1.0f + sig);
            }
        }

        // ---- local row max (reduce over 16 lanes of this half)
        float rmax[16];
#pragma unroll
        for (int i = 0; i < 16; ++i) {
            float v = sv[i];
            v = fmaxf(v, __shfl_xor(v, 1));
            v = fmaxf(v, __shfl_xor(v, 2));
            v = fmaxf(v, __shfl_xor(v, 4));
            v = fmaxf(v, __shfl_xor(v, 8));
            rmax[i] = v;
        }
        if (lm == 0) {
#pragma unroll
            for (int m01 = 0; m01 < 2; ++m01)
#pragma unroll
                for (int r = 0; r < 8; ++r)
                    wmaxs[wave][m01 * 16 + r + 8 * half] = rmax[m01 * 8 + r];
        }
        __syncthreads();

        // ---- new running max + accumulator rescale factor (per C-layout row)
        float mnew[16], scale[16];
#pragma unroll
        for (int m01 = 0; m01 < 2; ++m01) {
#pragma unroll
            for (int r = 0; r < 8; ++r) {
                int row = m01 * 16 + r + 8 * half;
                float mp = run_m[row];
                float m = mp;
#pragma unroll
                for (int w = 0; w < NWAVES; ++w) m = fmaxf(m, wmaxs[w][row]);
                mnew[m01 * 8 + r] = m;
                scale[m01 * 8 + r] = __expf(mp - m);
            }
        }

        // ---- P = exp(s - mnew) -> LDS (bf16), local row sums
        float rsum[16];
#pragma unroll
        for (int m01 = 0; m01 < 2; ++m01) {
#pragma unroll
            for (int r = 0; r < 8; ++r) {
                int i = m01 * 8 + r;
                float p = __expf(sv[i] - mnew[i]);
                Ps[m01 * 16 + r + 8 * half][wave * 16 + lm] = (__bf16)p;
                float v = p;
                v += __shfl_xor(v, 1);
                v += __shfl_xor(v, 2);
                v += __shfl_xor(v, 4);
                v += __shfl_xor(v, 8);
                rsum[i] = v;
            }
        }
        if (lm == 0) {
#pragma unroll
            for (int m01 = 0; m01 < 2; ++m01)
#pragma unroll
                for (int r = 0; r < 8; ++r)
                    wsums[wave][m01 * 16 + r + 8 * half] = rsum[m01 * 8 + r];
        }
        __syncthreads();

        // ---- one thread per row updates running stats
        if (tid < QROWS) {
            float mp = run_m[tid];
            float m = mp;
#pragma unroll
            for (int w = 0; w < NWAVES; ++w) m = fmaxf(m, wmaxs[w][tid]);
            float s = run_s[tid] * __expf(mp - m);
#pragma unroll
            for (int w = 0; w < NWAVES; ++w) s += wsums[w][tid];
            run_m[tid] = m;
            run_s[tid] = s;
        }

        // ---- rescale O accumulators
#pragma unroll
        for (int t = 0; t < 8; ++t)
#pragma unroll
            for (int m01 = 0; m01 < 2; ++m01)
#pragma unroll
                for (int r = 0; r < 8; ++r)
                    oacc[t][m01][r] *= scale[m01 * 8 + r];

        // ---- O += P @ V  (this wave's 128-wide D slice), K = KT.
        // Each V fragment is consumed by both M-tiles.
        v16bf pf[2][4];
#pragma unroll
        for (int m01 = 0; m01 < 2; ++m01)
#pragma unroll
            for (int k4 = 0; k4 < 4; ++k4)
                pf[m01][k4] = load_afrag(&Ps[m01 * 16][0] + k4 * 32, KT + 8, lane);

        const __bf16* Vrows =
            Vt + ((long long)b * D_MODEL + wave * 128) * SEQ + ktile * KT;
#pragma unroll
        for (int t = 0; t < 8; ++t) {
            const __bf16* vtp = Vrows + (long long)(t * 16) * SEQ;
#pragma unroll
            for (int k4 = 0; k4 < 4; ++k4) {
                v16bf bf_ = load_bfrag(vtp + k4 * 32, SEQ, lane);
                oacc[t][0] = wmma_bf16(pf[0][k4], bf_, oacc[t][0]);
                oacc[t][1] = wmma_bf16(pf[1][k4], bf_, oacc[t][1]);
            }
        }
        __syncthreads();  // protect Ps / wmaxs / wsums / run_* for next tile
    }

    // ---- normalize and store O (bf16) for the harmonization GEMM
    float inv[16];
#pragma unroll
    for (int m01 = 0; m01 < 2; ++m01)
#pragma unroll
        for (int r = 0; r < 8; ++r)
            inv[m01 * 8 + r] = 1.0f / run_s[m01 * 16 + r + 8 * half];
#pragma unroll
    for (int t = 0; t < 8; ++t) {
#pragma unroll
        for (int m01 = 0; m01 < 2; ++m01) {
#pragma unroll
            for (int r = 0; r < 8; ++r) {
                long long row = qrow0 + m01 * 16 + r + 8 * half;
                int col = wave * 128 + t * 16 + lm;
                Ob[row * D_MODEL + col] = (__bf16)(oacc[t][m01][r] * inv[m01 * 8 + r]);
            }
        }
    }
}

// ---------------------------------------------------------------------------
extern "C" void kernel_launch(void* const* d_in, const int* in_sizes, int n_in,
                              void* d_out, int out_size, void* d_ws, size_t ws_size,
                              hipStream_t stream) {
    const float* q     = (const float*)d_in[0];
    const float* k     = (const float*)d_in[1];
    const float* v     = (const float*)d_in[2];
    const float* Wgq   = (const float*)d_in[3];
    const float* bgq   = (const float*)d_in[4];
    const float* Wgk   = (const float*)d_in[5];
    const float* bgk   = (const float*)d_in[6];
    const float* wgate = (const float*)d_in[7];
    const float* bgate = (const float*)d_in[8];
    const float* Wh    = (const float*)d_in[9];
    const float* bh    = (const float*)d_in[10];
    float* out = (float*)d_out;

    const long long NQK = (long long)BATCH * SEQ * D_MODEL;  // 8388608
    const long long NG  = (long long)BATCH * SEQ * G_DIM;

    // workspace carve-up (256B aligned)
    char* ws = (char*)d_ws;
    size_t off = 0;
    auto carve = [&](size_t bytes) {
        char* p = ws + off;
        off = (off + bytes + 255) & ~(size_t)255;
        return p;
    };
    __bf16* Qbf   = (__bf16*)carve(NQK * 2);
    __bf16* Kbf   = (__bf16*)carve(NQK * 2);
    __bf16* Vt    = (__bf16*)carve(NQK * 2);
    __bf16* Obf   = (__bf16*)carve(NQK * 2);
    __bf16* Gqb   = (__bf16*)carve(NG * 2);
    __bf16* Gkb   = (__bf16*)carve(NG * 2);
    __bf16* Wgqb  = (__bf16*)carve((size_t)G_DIM * D_MODEL * 2);
    __bf16* Wgkb  = (__bf16*)carve((size_t)G_DIM * D_MODEL * 2);
    __bf16* Whb   = (__bf16*)carve((size_t)D_MODEL * D_MODEL * 2);
    (void)ws_size; (void)n_in; (void)in_sizes; (void)out_size;

    // 1) precision conversion
    k_cvt_bf16<<<4096, 256, 0, stream>>>(q,   Qbf,  NQK);
    k_cvt_bf16<<<4096, 256, 0, stream>>>(k,   Kbf,  NQK);
    k_cvt_bf16<<<256,  256, 0, stream>>>(Wgq, Wgqb, (long long)G_DIM * D_MODEL);
    k_cvt_bf16<<<256,  256, 0, stream>>>(Wgk, Wgkb, (long long)G_DIM * D_MODEL);
    k_cvt_bf16<<<4096, 256, 0, stream>>>(Wh,  Whb,  (long long)D_MODEL * D_MODEL);

    // 2) V -> Vt (transposed bf16)
    k_transpose_v<<<BATCH * (SEQ / 32) * (D_MODEL / 32), 256, 0, stream>>>(v, Vt);

    // 3) gnosis projections: gq = tanh(Q Wgq^T + bgq), gk = tanh(K Wgk^T + bgk)
    {
        int M = BATCH * SEQ, N = G_DIM, K = D_MODEL;
        int blocks = (M / 32) * (N / 64) / NWAVES;  // 32
        k_gemm_bf16<1, __bf16><<<blocks, 256, 0, stream>>>(Qbf, Wgqb, bgq, Gqb, M, N, K, N);
        k_gemm_bf16<1, __bf16><<<blocks, 256, 0, stream>>>(Kbf, Wgkb, bgk, Gkb, M, N, K, N);
    }

    // 4) fused gated attention -> Obf
    k_attention<<<BATCH * (SEQ / QROWS), 256, 0, stream>>>(Qbf, Kbf, Vt, Gqb, Gkb,
                                                           wgate, bgate, Obf);

    // 5) harmonization: out = O Wh^T + bh  (fp32 output)
    {
        int M = BATCH * SEQ, N = D_MODEL, K = D_MODEL;
        int blocks = (M / 32) * (N / 64) / NWAVES;  // 512
        k_gemm_bf16<0, float><<<blocks, 256, 0, stream>>>(Obf, Whb, bh, out, M, N, K, N);
    }
}
